// VanillaGNO_61177514164376
// MI455X (gfx1250) — matrix operand order repaired and checked by reading, not compiled, not run on hardware
//
#include <hip/hip_runtime.h>
#include <hip/hip_bf16.h>
#include <math.h>

// ---------------- problem constants (match reference) ----------------
#define N_GRAPHS   8
#define NODES_PG   1024
#define EDGES_PG   2048
#define NN         (N_GRAPHS * NODES_PG)   // 8192 nodes
#define EE         (N_GRAPHS * EDGES_PG)   // 16384 edges
#define IN_DIM     10
#define EDGE_DIM   9
#define HID        64
#define OUT_DIM    3
#define TLAYERS    5
#define KTOT       4160                    // 65*64: 64 h-rows + 1 bias row (h==1), each x64 j

typedef __attribute__((ext_vector_type(16))) __bf16 v16bf;
typedef __attribute__((ext_vector_type(8)))  __bf16 v8bf;
typedef __attribute__((ext_vector_type(8)))  float  v8f;
typedef __attribute__((ext_vector_type(4)))  float  v4f;
typedef __attribute__((ext_vector_type(4)))  unsigned int v4u;
typedef __attribute__((ext_vector_type(4)))  int v4i;
typedef __attribute__((ext_vector_type(8)))  int v8i;

#if defined(__gfx1250__) && __has_builtin(__builtin_amdgcn_tensor_load_to_lds)
#define HAVE_TDM 1
#else
#define HAVE_TDM 0
#endif

// ---------------- utility kernels ----------------
__global__ __launch_bounds__(256) void k_zero(float* __restrict__ p, int n) {
  int i = blockIdx.x * 256 + threadIdx.x;
  if (i < n) p[i] = 0.0f;
}

__global__ __launch_bounds__(256) void k_count_deg(const long long* __restrict__ dst,
                                                   float* __restrict__ deg) {
  int e = blockIdx.x * 256 + threadIdx.x;
  if (e < EE) atomicAdd(&deg[dst[e]], 1.0f);
}

// Per-graph standardization of edge_attr (edges are block-structured per graph).
__global__ __launch_bounds__(256) void k_edge_stats(const float* __restrict__ ea,
                                                    float* __restrict__ ean) {
  __shared__ float s_sum[EDGE_DIM], s_sq[EDGE_DIM], s_mean[EDGE_DIM], s_istd[EDGE_DIM];
  const int g = blockIdx.x, tid = threadIdx.x;
  if (tid < EDGE_DIM) { s_sum[tid] = 0.0f; s_sq[tid] = 0.0f; }
  __syncthreads();
  float ps[EDGE_DIM], pq[EDGE_DIM];
#pragma unroll
  for (int d = 0; d < EDGE_DIM; ++d) { ps[d] = 0.0f; pq[d] = 0.0f; }
  for (int e = tid; e < EDGES_PG; e += 256) {
    const float* r = ea + ((size_t)g * EDGES_PG + e) * EDGE_DIM;
#pragma unroll
    for (int d = 0; d < EDGE_DIM; ++d) { float v = r[d]; ps[d] += v; pq[d] += v * v; }
  }
#pragma unroll
  for (int d = 0; d < EDGE_DIM; ++d) { atomicAdd(&s_sum[d], ps[d]); atomicAdd(&s_sq[d], pq[d]); }
  __syncthreads();
  if (tid < EDGE_DIM) {
    float m   = s_sum[tid] / (float)EDGES_PG;
    float var = fmaxf(s_sq[tid] / (float)EDGES_PG - m * m, 0.0f);
    s_mean[tid] = m;
    s_istd[tid] = 1.0f / (sqrtf(var) + 1e-6f);
  }
  __syncthreads();
  for (int idx = tid; idx < EDGES_PG * EDGE_DIM; idx += 256) {
    int d = idx % EDGE_DIM;
    size_t gi = (size_t)g * EDGES_PG * EDGE_DIM + idx;
    ean[gi] = (ea[gi] - s_mean[d]) * s_istd[d];
  }
}

// Generic dense layer: out[r,c] = act(in[r,:] @ W[:,c] + b[c]). Tiny VALU work.
__global__ __launch_bounds__(256) void k_linear(const float* __restrict__ in,
                                                const float* __restrict__ W,
                                                const float* __restrict__ b,
                                                float* __restrict__ out,
                                                int rows, int K, int C, int relu) {
  int idx = blockIdx.x * 256 + threadIdx.x;
  if (idx >= rows * C) return;
  int r = idx / C, c = idx % C;
  const float* ir = in + (size_t)r * K;
  float acc = b[c];
  for (int k = 0; k < K; ++k) acc = fmaf(ir[k], W[(size_t)k * C + c], acc);
  out[idx] = relu ? fmaxf(acc, 0.0f) : acc;
}

// Repack layer-t k_w3 [64 m][4096 (i*64+j)] + k_b3 [4096] into bf16 W3T[i][k],
// k = m*64+j for k<4096, bias row (h==1) at k in [4096,4160).
__global__ __launch_bounds__(256) void k_w3_prep(const float* __restrict__ kw3,
                                                 const float* __restrict__ kb3,
                                                 __bf16* __restrict__ W3T) {
  int idx = blockIdx.x * 256 + threadIdx.x;
  if (idx >= HID * KTOT) return;
  int i = idx / KTOT;
  int k = idx % KTOT;
  float val;
  if (k < HID * HID) {
    int m = k >> 6, j = k & 63;
    val = kw3[(size_t)m * (HID * HID) + i * HID + j];
  } else {
    int j = k - HID * HID;
    val = kb3[i * HID + j];
  }
  W3T[idx] = (__bf16)val;
}

// ---------------- the WMMA message kernel ----------------
// msg = (h (x)_row v[src]) @ W3' accumulated into aggr[dst] via atomics.
// 256-thread block = 8 waves; each wave computes a 16-edge x 64-col strip.
// h tile staged to LDS via TDM (padded rows); v[src] held entirely in registers.

#define WMMA_BF16(A, B, C) \
  __builtin_amdgcn_wmma_f32_16x16x32_bf16(false, (A), false, (B), (short)0, (C), false, false)

#define LD16(Bv, bp) {                                   \
    v8bf t0_ = *(const v8bf*)(bp);                       \
    v8bf t1_ = *(const v8bf*)((bp) + 16);                \
    _Pragma("unroll")                                    \
    for (int p_ = 0; p_ < 8; ++p_) { Bv[p_] = t0_[p_]; Bv[p_ + 8] = t1_[p_]; } }

#define LOAD_B4(kb, B0, B1, B2, B3) {                    \
    LD16(B0, Wl + 0 * (16 * KTOT) + (kb));               \
    LD16(B1, Wl + 1 * (16 * KTOT) + (kb));               \
    LD16(B2, Wl + 2 * (16 * KTOT) + (kb));               \
    LD16(B3, Wl + 3 * (16 * KTOT) + (kb)); }

#define BUILD_A(Av, hs, Ra, Rb, Rc, Rd) {                \
    _Pragma("unroll")                                    \
    for (int p_ = 0; p_ < 4; ++p_) {                     \
      Av[p_]      = (__bf16)((hs) * (Ra)[p_]);           \
      Av[p_ + 4]  = (__bf16)((hs) * (Rb)[p_]);           \
      Av[p_ + 8]  = (__bf16)((hs) * (Rc)[p_]);           \
      Av[p_ + 12] = (__bf16)((hs) * (Rd)[p_]); } }

__global__ __launch_bounds__(256) void k_msg_wmma(const float* __restrict__ h,    // [E][64]
                                                  const float* __restrict__ v,    // [N][64]
                                                  const long long* __restrict__ src,
                                                  const long long* __restrict__ dst,
                                                  const __bf16* __restrict__ W3T, // [64][KTOT]
                                                  float* __restrict__ aggr) {     // [N][64]
  __shared__ float s_h[128 * 65 + 8];  // 65-dword row stride: bank-conflict free
  const int tid = threadIdx.x;
  const int ebase = blockIdx.x * 128;

#if HAVE_TDM
  // TDM: async DMA of the contiguous 128x64 f32 h-tile into LDS, inserting a
  // 1-dword pad every 64 dwords (row stride 65). Issued once by wave 0.
  if (tid < 32) {
    unsigned long long ga = (unsigned long long)(uintptr_t)(h + (size_t)ebase * HID);
    unsigned lds_off = (unsigned)(uintptr_t)(&s_h[0]);  // low 32 bits = LDS offset
    v4u g0;
    g0[0] = 1u;                                              // count=1, user descriptor
    g0[1] = lds_off;                                         // lds_addr
    g0[2] = (unsigned)(ga & 0xFFFFFFFFu);                    // global_addr[31:0]
    g0[3] = (unsigned)((ga >> 32) & 0x01FFFFFFu) | (2u << 30); // addr[56:32] | type=2
    v8i g1;
    g1[0] = (2 << 16) | (1 << 20) | (5 << 22);  // data_size=4B, pad_enable, every 64dw, +1dw
    g1[1] = (int)(0x2000u << 16);               // tensor_dim0 = 8192 (lo16 @ bits 63:48)
    g1[2] = (int)(1u << 16);                    // tensor_dim0 hi16=0 | tensor_dim1 = 1
    g1[3] = (int)(0x2000u << 16);               // tensor_dim1 hi16=0 | tile_dim0 = 8192
    g1[4] = 1;                                  // tile_dim1 = 1, tile_dim2 = 0
    g1[5] = 8192;                               // tensor_dim0_stride lo32
    g1[6] = (int)(0x2000u << 16);               // dim0_stride hi16=0 | dim1_stride lo16
    g1[7] = 0;                                  // dim1_stride hi32
    v4i gz = {0, 0, 0, 0};
#if defined(__clang_major__) && (__clang_major__ >= 23)
    v8i gz8 = {0, 0, 0, 0, 0, 0, 0, 0};
    __builtin_amdgcn_tensor_load_to_lds(g0, g1, gz, gz, gz8, 0);
#else
    __builtin_amdgcn_tensor_load_to_lds(g0, g1, gz, gz, 0);
#endif
    __builtin_amdgcn_s_wait_tensorcnt(0);
  }
#else
  for (int idx = tid; idx < 128 * HID; idx += 256) {
    int le = idx >> 6, c = idx & 63;
    s_h[le * 65 + c] = h[(size_t)(ebase + le) * HID + c];
  }
#endif
  __syncthreads();

  const int wave = tid >> 5;         // strip: edges [wave*16, wave*16+16)
  const int lane = tid & 31;
  const int half = lane >> 4;        // K-half selector for A/B fragments
  const int col  = lane & 15;        // A row (M) / B+D column (N)
  const int le   = wave * 16 + col;

  // v[src] for this lane: only 32 of 64 floats are ever touched -> registers.
  const long long se = src[ebase + le];
  const float* vrow = v + (size_t)se * HID;
  const int jo = 8 * half;
  v4f r0a = *(const v4f*)(vrow + jo);            // jbase=0,  run0: j = jo..jo+7
  v4f r0b = *(const v4f*)(vrow + jo + 4);
  v4f r1a = *(const v4f*)(vrow + jo + 16);       // jbase=0,  run1: j = jo+16..jo+23
  v4f r1b = *(const v4f*)(vrow + jo + 20);
  v4f r2a = *(const v4f*)(vrow + 32 + jo);       // jbase=32, run0
  v4f r2b = *(const v4f*)(vrow + 32 + jo + 4);
  v4f r3a = *(const v4f*)(vrow + 48 + jo);       // jbase=32, run1
  v4f r3b = *(const v4f*)(vrow + 48 + jo + 4);

  const __bf16* Wl = W3T + (size_t)col * KTOT + 8 * half;  // B lane base

  v8f acc0 = {}, acc1 = {}, acc2 = {}, acc3 = {};
  v16bf be0, be1, be2, be3, bo0, bo1, bo2, bo3;

  LOAD_B4(0, be0, be1, be2, be3);  // prime even-slab B of m=0

  for (int m = 0; m <= 64; ++m) {   // m==64 is the bias row (h == 1)
    const float hs = (m < 64) ? s_h[le * 65 + m] : 1.0f;
    v16bf ae, ao;
    BUILD_A(ae, hs, r0a, r0b, r1a, r1b);
    BUILD_A(ao, hs, r2a, r2b, r3a, r3b);

    LOAD_B4(m * 64 + 32, bo0, bo1, bo2, bo3);   // odd slab B (covers during even WMMAs)
    acc0 = WMMA_BF16(ae, be0, acc0);
    acc1 = WMMA_BF16(ae, be1, acc1);
    acc2 = WMMA_BF16(ae, be2, acc2);
    acc3 = WMMA_BF16(ae, be3, acc3);

    const int mn = (m < 64) ? (m + 1) : 64;     // clamp: harmless reload on last iter
    LOAD_B4(mn * 64, be0, be1, be2, be3);       // next even slab B (covers odd WMMAs)
    acc0 = WMMA_BF16(ao, bo0, acc0);
    acc1 = WMMA_BF16(ao, bo1, acc1);
    acc2 = WMMA_BF16(ao, bo2, acc2);
    acc3 = WMMA_BF16(ao, bo3, acc3);
  }

  // Scatter-add C/D tile: VGPR r -> row M = r + 8*half, column N = col.
#pragma unroll
  for (int r = 0; r < 8; ++r) {
    int M = r + 8 * half;
    long long e = ebase + wave * 16 + M;
    float* ap = aggr + (size_t)dst[e] * HID;
    atomicAdd(ap +  0 + col, acc0[r]);
    atomicAdd(ap + 16 + col, acc1[r]);
    atomicAdd(ap + 32 + col, acc2[r]);
    atomicAdd(ap + 48 + col, acc3[r]);
  }
}

// v' = relu(v @ root_w + root_b + aggr/deg)
__global__ __launch_bounds__(256) void k_node_update(const float* __restrict__ v,
                                                     const float* __restrict__ rw,
                                                     const float* __restrict__ rb,
                                                     const float* __restrict__ aggr,
                                                     const float* __restrict__ deg,
                                                     float* __restrict__ vout) {
  int idx = blockIdx.x * 256 + threadIdx.x;
  if (idx >= NN * HID) return;
  int n = idx >> 6, c = idx & 63;
  const float* vr = v + (size_t)n * HID;
  float acc = rb[c] + aggr[idx] / fmaxf(deg[n], 1.0f);
  for (int k = 0; k < HID; ++k) acc = fmaf(vr[k], rw[k * HID + c], acc);
  vout[idx] = fmaxf(acc, 0.0f);
}

// ---------------- launch ----------------
extern "C" void kernel_launch(void* const* d_in, const int* in_sizes, int n_in,
                              void* d_out, int out_size, void* d_ws, size_t ws_size,
                              hipStream_t stream) {
  const float* x         = (const float*)d_in[0];
  const float* edge_attr = (const float*)d_in[1];
  const float* lift_w1   = (const float*)d_in[2];
  const float* lift_b1   = (const float*)d_in[3];
  const float* lift_w2   = (const float*)d_in[4];
  const float* lift_b2   = (const float*)d_in[5];
  const float* root_w    = (const float*)d_in[6];
  const float* root_b    = (const float*)d_in[7];
  const float* k_w1      = (const float*)d_in[8];
  const float* k_b1      = (const float*)d_in[9];
  const float* k_w2      = (const float*)d_in[10];
  const float* k_b2      = (const float*)d_in[11];
  const float* k_w3      = (const float*)d_in[12];
  const float* k_b3      = (const float*)d_in[13];
  const float* proj_w1   = (const float*)d_in[14];
  const float* proj_b1   = (const float*)d_in[15];
  const float* proj_w2   = (const float*)d_in[16];
  const float* proj_b2   = (const float*)d_in[17];
  const long long* eidx  = (const long long*)d_in[18]; // int64 [2,E]
  float* out = (float*)d_out;

  // workspace layout (floats; W3T bf16 tail stays 16B-aligned)
  float* ws   = (float*)d_ws;
  size_t o    = 0;
  float* ea_n = ws + o; o += (size_t)EE * EDGE_DIM;
  float* deg  = ws + o; o += NN;
  float* t1   = ws + o; o += (size_t)NN * HID;   // lift/proj temp
  float* vA   = ws + o; o += (size_t)NN * HID;
  float* vB   = ws + o; o += (size_t)NN * HID;
  float* h1   = ws + o; o += (size_t)EE * HID;
  float* hb   = ws + o; o += (size_t)EE * HID;
  float* aggr = ws + o; o += (size_t)NN * HID;
  __bf16* W3T = (__bf16*)(ws + o);

  const long long* src = eidx;
  const long long* dst = eidx + EE;

  k_edge_stats<<<N_GRAPHS, 256, 0, stream>>>(edge_attr, ea_n);
  k_zero<<<(NN + 255) / 256, 256, 0, stream>>>(deg, NN);
  k_count_deg<<<(EE + 255) / 256, 256, 0, stream>>>(dst, deg);

  // lift MLP
  k_linear<<<(NN * HID + 255) / 256, 256, 0, stream>>>(x, lift_w1, lift_b1, t1, NN, IN_DIM, HID, 1);
  k_linear<<<(NN * HID + 255) / 256, 256, 0, stream>>>(t1, lift_w2, lift_b2, vA, NN, HID, HID, 0);

  float* vcur = vA;
  float* vnxt = vB;
  for (int t = 0; t < TLAYERS; ++t) {
    k_linear<<<(EE * HID + 255) / 256, 256, 0, stream>>>(
        ea_n, k_w1 + (size_t)t * EDGE_DIM * HID, k_b1 + t * HID, h1, EE, EDGE_DIM, HID, 1);
    k_linear<<<(EE * HID + 255) / 256, 256, 0, stream>>>(
        h1, k_w2 + (size_t)t * HID * HID, k_b2 + t * HID, hb, EE, HID, HID, 1);
    k_w3_prep<<<(HID * KTOT + 255) / 256, 256, 0, stream>>>(
        k_w3 + (size_t)t * HID * HID * HID, k_b3 + (size_t)t * HID * HID, W3T);
    k_zero<<<(NN * HID + 255) / 256, 256, 0, stream>>>(aggr, NN * HID);
    k_msg_wmma<<<EE / 128, 256, 0, stream>>>(hb, vcur, src, dst, W3T, aggr);
    k_node_update<<<(NN * HID + 255) / 256, 256, 0, stream>>>(
        vcur, root_w + (size_t)t * HID * HID, root_b + t * HID, aggr, deg, vnxt);
    float* tmp = vcur; vcur = vnxt; vnxt = tmp;
  }

  // projection
  k_linear<<<(NN * HID + 255) / 256, 256, 0, stream>>>(vcur, proj_w1, proj_b1, t1, NN, HID, HID, 1);
  k_linear<<<(NN * OUT_DIM + 255) / 256, 256, 0, stream>>>(t1, proj_w2, proj_b2, out, NN, HID, OUT_DIM, 0);
}
